// SchNetInteractionTriple_85658827752014
// MI455X (gfx1250) — matrix operand order repaired and checked by reading, not compiled, not run on hardware
//
#include <hip/hip_runtime.h>
#include <hip/hip_bf16.h>

typedef __attribute__((ext_vector_type(16))) _Float16 v16h;
typedef __attribute__((ext_vector_type(8)))  float    v8f;
typedef __attribute__((ext_vector_type(4)))  int      v4i;

#define B_    8
#define AT_   96
#define NBR_  512
#define NANG_ 125
#define NF_   128
#define NB_   128

// dynamic LDS layout: w1 (32KB) | w2 (32KB) | hbuf 8 waves x 4KB (32KB)
#define LDS_W_BYTES   (64 * 1024)
#define LDS_DYN_BYTES (96 * 1024)

__device__ __forceinline__ float ssp(float x) {
    // shifted softplus: softplus(x) - ln(2), overflow-safe
    float sp = (x > 20.0f) ? x : log1pf(expf(x));
    return sp - 0.69314718055994531f;
}

// ---------------------------------------------------------------------------
// Pack W1 (125x128, zero-padded to 128 in K) and W2 (128x128) into the WMMA
// B-operand image for v_wmma_f32_16x16x32_f16, in this kernel's contiguous k order (A matches):
//   lane l: n = (l & 15); K-half = (l >> 4) selects K 0-15 vs 16-31 of chunk
//   vgpr i (0..7): k = chunk*32 + (l>>4)*16 + i*2 + {0,1}
// Stored as [chunk(4)][ntile(8)][lane(32)][elem(16)] f16 => 16384 f16 each,
// w1p and w2p contiguous in workspace (64KB total).
// ---------------------------------------------------------------------------
__global__ void pack_weights_kernel(const float* __restrict__ W1,
                                    const float* __restrict__ W2,
                                    _Float16* __restrict__ w1p,
                                    _Float16* __restrict__ w2p) {
    int idx = blockIdx.x * blockDim.x + threadIdx.x;   // 0 .. 32767
    if (idx >= 2 * 16384) return;
    int which = idx >> 14;
    int e     = idx & 16383;              // ch<<12 | t<<9 | lane<<4 | elem
    int elem  = e & 15;
    int lane  = (e >> 4) & 31;
    int t     = (e >> 9) & 7;
    int ch    = (e >> 12) & 3;
    int i  = elem >> 1;
    int lo = elem & 1;
    int k  = ch * 32 + (lane >> 4) * 16 + i * 2 + lo;
    int n  = (lane & 15) + 16 * t;
    if (which == 0) {
        float v = (k < NANG_) ? W1[k * NF_ + n] : 0.0f;
        w1p[e] = (_Float16)v;
    } else {
        w2p[e] = (_Float16)W2[k * NF_ + n];
    }
}

// ---------------------------------------------------------------------------
// y = x @ Win  : (768 x 128) @ (128 x 128), one block per (b, atom)
// ---------------------------------------------------------------------------
__global__ __launch_bounds__(128) void compute_y_kernel(
        const float* __restrict__ x, const float* __restrict__ Win,
        float* __restrict__ y) {
    __shared__ float xr[NB_];
    int ba = blockIdx.x;
    int f  = threadIdx.x;
    xr[f] = x[(size_t)ba * NB_ + f];
    __syncthreads();
    float s = 0.0f;
    #pragma unroll 4
    for (int k = 0; k < NB_; ++k) s += xr[k] * Win[k * NF_ + f];
    y[(size_t)ba * NF_ + f] = s;
}

// ---------------------------------------------------------------------------
// Fused: filternet (2x WMMA GEMM) + gather-multiply-reduce + f2out + dense +
// residual. One block per (b, atom); 8 waves; each wave owns 4 row tiles of
// 16 triples. Packed weights staged to LDS once per block via async-to-LDS.
// ---------------------------------------------------------------------------
__global__ __launch_bounds__(256) void triple_interaction_kernel(
        const float* __restrict__ x,
        const float* __restrict__ triple,
        const int*   __restrict__ nj,
        const int*   __restrict__ nk,
        const float* __restrict__ mask,
        const _Float16* __restrict__ wpacked,   // w1p|w2p contiguous, 64KB
        const float* __restrict__ b1,
        const float* __restrict__ b2,
        const float* __restrict__ yglob,
        const float* __restrict__ wf2,
        const float* __restrict__ bf2,
        const float* __restrict__ wd,
        const float* __restrict__ bd,
        float* __restrict__ out) {
    extern __shared__ char smem_dyn[];
    _Float16* w1l   = (_Float16*)(smem_dyn);
    _Float16* w2l   = (_Float16*)(smem_dyn + 32 * 1024);
    _Float16* hbase = (_Float16*)(smem_dyn + LDS_W_BYTES);

    __shared__ float agg[NF_];
    __shared__ float vbuf[NB_];

    const int ba   = blockIdx.x;             // 0..767  (b*AT + a)
    const int b    = ba / AT_;
    const int tid  = threadIdx.x;
    const int wave = tid >> 5;
    const int lane = tid & 31;
    const int lm   = lane & 15;
    const int kh   = lane >> 4;

    if (tid < NF_) agg[tid] = 0.0f;

    // ---- stage packed weights (64KB) into LDS, async-to-LDS if available --
    {
        const char* gsrc = (const char*)wpacked;
        #pragma unroll
        for (int it = 0; it < LDS_W_BYTES / (256 * 16); ++it) {
            int ofs = (it * 256 + tid) * 16;
#if __has_builtin(__builtin_amdgcn_global_load_async_to_lds_b128)
            __builtin_amdgcn_global_load_async_to_lds_b128(
                (__attribute__((address_space(1))) v4i*)(gsrc + ofs),
                (__attribute__((address_space(3))) v4i*)(smem_dyn + ofs),
                0, 0);
#else
            *(int4*)(smem_dyn + ofs) = *(const int4*)(gsrc + ofs);
#endif
        }
#if __has_builtin(__builtin_amdgcn_global_load_async_to_lds_b128)
 #if __has_builtin(__builtin_amdgcn_s_wait_asynccnt)
        __builtin_amdgcn_s_wait_asynccnt(0);
 #else
        asm volatile("s_wait_asynccnt 0" ::: "memory");
 #endif
#endif
    }
    __syncthreads();

    const float* yb      = yglob + (size_t)b * AT_ * NF_;
    const float* trip_ba = triple + (size_t)ba * NBR_ * NANG_;
    const int*   njb     = nj + (size_t)ba * NBR_;
    const int*   nkb     = nk + (size_t)ba * NBR_;
    const float* mb      = mask + (size_t)ba * NBR_;

    float b1v[8], b2v[8], accv[8];
    #pragma unroll
    for (int t = 0; t < 8; ++t) {
        b1v[t] = b1[lm + 16 * t];
        b2v[t] = b2[lm + 16 * t];
        accv[t] = 0.0f;
    }

    _Float16* hl = hbase + wave * (16 * NF_);
    const v16h* w1v = (const v16h*)w1l;
    const v16h* w2v = (const v16h*)w2l;

    for (int rt = wave; rt < 32; rt += 8) {
        const int row0 = rt * 16;
        const float* rp = trip_ba + (size_t)(row0 + lm) * NANG_;

        // prefetch next row tile of triple features (global_prefetch_b8)
        if (rt + 8 < 32) {
            const char* pf = (const char*)(trip_ba + (size_t)(row0 + 128 + lm) * NANG_);
            __builtin_prefetch(pf, 0, 1);
            __builtin_prefetch(pf + 256, 0, 1);
        }

        // ---------------- layer 1: h = ssp(T @ W1 + b1) -------------------
        v8f c1[8];
        #pragma unroll
        for (int t = 0; t < 8; ++t) c1[t] = (v8f){0.f,0.f,0.f,0.f,0.f,0.f,0.f,0.f};

        #pragma unroll
        for (int ch = 0; ch < 4; ++ch) {
            v16h av;
            #pragma unroll
            for (int i = 0; i < 8; ++i) {
                // A layout: k = ch*32 + kh*16 + i*2
                int k = ch * 32 + kh * 16 + i * 2;
                float f0 = (k     < NANG_) ? rp[k]     : 0.0f;
                float f1 = (k + 1 < NANG_) ? rp[k + 1] : 0.0f;
                av[2 * i]     = (_Float16)f0;
                av[2 * i + 1] = (_Float16)f1;
            }
            #pragma unroll
            for (int t = 0; t < 8; ++t) {
                v16h bv = w1v[(ch * 8 + t) * 32 + lane];   // ds_load_b128 x2
                c1[t] = __builtin_amdgcn_wmma_f32_16x16x32_f16(
                            false, av, false, bv, (short)0, c1[t], false, false);
            }
        }

        // bias + ssp, transpose C/D layout -> A layout via per-wave LDS tile
        #pragma unroll
        for (int t = 0; t < 8; ++t) {
            int n = lm + 16 * t;
            #pragma unroll
            for (int r = 0; r < 8; ++r) {
                int m = r + 8 * kh;
                hl[m * NF_ + n] = (_Float16)ssp(c1[t][r] + b1v[t]);
            }
        }
        // per-wave LDS RAW: DS ops are in-order per wave; wait + compiler fence
        asm volatile("s_wait_dscnt 0" ::: "memory");

        // ---------------- layer 2: Wt = h @ W2 + b2 -----------------------
        v8f c2[8];
        #pragma unroll
        for (int t = 0; t < 8; ++t) c2[t] = (v8f){0.f,0.f,0.f,0.f,0.f,0.f,0.f,0.f};

        #pragma unroll
        for (int ch = 0; ch < 4; ++ch) {
            v16h av;
            #pragma unroll
            for (int i = 0; i < 8; ++i) {
                int k = ch * 32 + kh * 16 + i * 2;
                av[2 * i]     = hl[lm * NF_ + k];
                av[2 * i + 1] = hl[lm * NF_ + k + 1];
            }
            #pragma unroll
            for (int t = 0; t < 8; ++t) {
                v16h bv = w2v[(ch * 8 + t) * 32 + lane];
                c2[t] = __builtin_amdgcn_wmma_f32_16x16x32_f16(
                            false, av, false, bv, (short)0, c2[t], false, false);
            }
        }

        // ------- continuous-filter conv: acc[n] += Wt[m][n]*yj*yk*mask ----
        #pragma unroll
        for (int r = 0; r < 8; ++r) {
            int m   = r + 8 * kh;
            int row = row0 + m;
            int j   = njb[row];
            int kx  = nkb[row];
            float msk = mb[row];
            const float* yj = yb + (size_t)j  * NF_;
            const float* yk = yb + (size_t)kx * NF_;
            #pragma unroll
            for (int t = 0; t < 8; ++t) {
                int n = lm + 16 * t;
                float w = c2[t][r] + b2v[t];
                accv[t] += w * (yj[n] * yk[n] * msk);
            }
        }
    }

    // reduce per-lane partials into shared agg[128] (ds_add_f32)
    #pragma unroll
    for (int t = 0; t < 8; ++t) atomicAdd(&agg[lm + 16 * t], accv[t]);
    __syncthreads();

    // ---------------- f2out (ssp) -> dense -> residual --------------------
    if (tid < NB_) {
        float s = 0.0f;
        #pragma unroll 4
        for (int k = 0; k < NF_; ++k) s += agg[k] * wf2[k * NB_ + tid];
        vbuf[tid] = ssp(s + bf2[tid]);
    }
    __syncthreads();
    if (tid < NB_) {
        float s = 0.0f;
        #pragma unroll 4
        for (int f = 0; f < NB_; ++f) s += vbuf[f] * wd[f * NB_ + tid];
        out[(size_t)ba * NB_ + tid] = x[(size_t)ba * NB_ + tid] + s + bd[tid];
    }
}

// ---------------------------------------------------------------------------
extern "C" void kernel_launch(void* const* d_in, const int* in_sizes, int n_in,
                              void* d_out, int out_size, void* d_ws, size_t ws_size,
                              hipStream_t stream) {
    const float* x      = (const float*)d_in[0];
    const float* triple = (const float*)d_in[1];
    const int*   nj     = (const int*)  d_in[2];
    const int*   nk     = (const int*)  d_in[3];
    const float* mask   = (const float*)d_in[4];
    const float* W1     = (const float*)d_in[5];
    const float* b1     = (const float*)d_in[6];
    const float* W2     = (const float*)d_in[7];
    const float* b2     = (const float*)d_in[8];
    const float* Win    = (const float*)d_in[9];
    const float* Wf2    = (const float*)d_in[10];
    const float* bf2    = (const float*)d_in[11];
    const float* Wd     = (const float*)d_in[12];
    const float* bd     = (const float*)d_in[13];
    float* out = (float*)d_out;

    // workspace layout: w1p (32KB) | w2p (32KB, contiguous) | y (384KB)
    char* ws = (char*)d_ws;
    _Float16* w1p = (_Float16*)(ws);
    _Float16* w2p = (_Float16*)(ws + 32 * 1024);
    float*    y   = (float*)   (ws + 64 * 1024);

    (void)hipFuncSetAttribute(
        reinterpret_cast<const void*>(triple_interaction_kernel),
        hipFuncAttributeMaxDynamicSharedMemorySize, LDS_DYN_BYTES);

    pack_weights_kernel<<<128, 256, 0, stream>>>(W1, W2, w1p, w2p);
    compute_y_kernel<<<B_ * AT_, 128, 0, stream>>>(x, Win, y);
    triple_interaction_kernel<<<B_ * AT_, 256, LDS_DYN_BYTES, stream>>>(
        x, triple, nj, nk, mask, w1p, b1, b2, y, Wf2, bf2, Wd, bd, out);
}
